// OBBDinoTransformerDecoderLayer_45440753991920
// MI455X (gfx1250) — compile-verified, matrix operand order checked by reading, not run on hardware
//
#include <hip/hip_runtime.h>
#include <hip/hip_bf16.h>
#include <hip/hip_fp16.h>

// ---------------- problem constants ----------------
static constexpr int Dm   = 256;
static constexpr int Hh   = 8;
static constexpr int Pp   = 5;
static constexpr int NQq  = 900;
static constexpr int Bb   = 8;
static constexpr int DFFf = 1024;
static constexpr int Ss   = 21760;
static constexpr int Mrows = NQq * Bb;          // 7200 token rows (m = nq*B + b)

typedef __attribute__((ext_vector_type(16))) _Float16 v16h;
typedef __attribute__((ext_vector_type(8)))  float    v8f;

union Frag16 {
  v16h v;
  uint4 q[2];
  unsigned u[8];
  _Float16 h[16];
};

#define WMMA_F16(A, B, C) \
  __builtin_amdgcn_wmma_f32_16x16x32_f16(false, (A), false, (B), (short)0, (C), false, false)

// ---------------- packing / conversion kernels ----------------

__global__ void pack_q_kernel(const float* __restrict__ tgt, const float* __restrict__ pos,
                              _Float16* __restrict__ qh, _Float16* __restrict__ th, int n) {
  int i = blockIdx.x * blockDim.x + threadIdx.x;
  if (i < n) {
    float t = tgt[i];
    th[i] = (_Float16)t;
    qh[i] = (_Float16)(t + pos[i]);
  }
}

__global__ void cvt_kernel(const float* __restrict__ src, _Float16* __restrict__ dst, int n) {
  int i = blockIdx.x * blockDim.x + threadIdx.x;
  if (i < n) dst[i] = (_Float16)src[i];
}

// memory [S,B,256] f32 -> mem_h [B,S,256] f16 ; one thread = 4 contiguous channels
__global__ void memtr_kernel(const float* __restrict__ mem, _Float16* __restrict__ dst) {
  int i = blockIdx.x * blockDim.x + threadIdx.x;           // over S*B*64
  int sb = i >> 6;
  if (sb >= Ss * Bb) return;
  int c4 = (i & 63) * 4;
  int bI = sb % Bb, sI = sb / Bb;
  float4 v4 = *(const float4*)(mem + (size_t)sb * 256 + c4);
  size_t o = ((size_t)bI * Ss + sI) * 256 + c4;
  dst[o + 0] = (_Float16)v4.x;
  dst[o + 1] = (_Float16)v4.y;
  dst[o + 2] = (_Float16)v4.z;
  dst[o + 3] = (_Float16)v4.w;
}

// ---------------- generic WMMA GEMM: C[M,N] = A[M,K] @ W[N,K]^T + bias ----------------
// 256 threads = 8 waves, waves 4(M) x 2(N), wave tile 32x32, block tile 128x64.

__global__ __launch_bounds__(256) void gemm_f16(const _Float16* __restrict__ A,
                                                const _Float16* __restrict__ W,
                                                const float* __restrict__ bias,
                                                float* __restrict__ Cf,
                                                _Float16* __restrict__ Ch,
                                                int M, int N, int K, int relu) {
  const int lane = threadIdx.x & 31;
  const int wid  = threadIdx.x >> 5;
  const int r    = lane & 15;
  const int half = lane >> 4;
  const int wm = wid & 3, wn = wid >> 2;
  int row0 = blockIdx.x * 128 + wm * 32;
  int col0 = blockIdx.y * 64  + wn * 32;
  if (row0 >= M || col0 >= N) return;           // no barriers -> early wave exit is safe

  v8f acc[2][2] = {{{}, {}}, {{}, {}}};
  const _Float16* a0p = A + (size_t)(row0 + r) * K + half * 8;
  const _Float16* a1p = a0p + (size_t)16 * K;
  const _Float16* b0p = W + (size_t)(col0 + r) * K + half * 8;
  const _Float16* b1p = b0p + (size_t)16 * K;

  for (int kb = 0; kb < K; kb += 32) {
    if (kb + 32 < K) {                          // hint next k-tile -> global_prefetch_b8
      __builtin_prefetch(a0p + kb + 32, 0, 3);
      __builtin_prefetch(b0p + kb + 32, 0, 3);
    }
    Frag16 a0, a1, b0, b1;
    a0.q[0] = *(const uint4*)(a0p + kb); a0.q[1] = *(const uint4*)(a0p + kb + 16);
    a1.q[0] = *(const uint4*)(a1p + kb); a1.q[1] = *(const uint4*)(a1p + kb + 16);
    b0.q[0] = *(const uint4*)(b0p + kb); b0.q[1] = *(const uint4*)(b0p + kb + 16);
    b1.q[0] = *(const uint4*)(b1p + kb); b1.q[1] = *(const uint4*)(b1p + kb + 16);
    acc[0][0] = WMMA_F16(a0.v, b0.v, acc[0][0]);
    acc[0][1] = WMMA_F16(a0.v, b1.v, acc[0][1]);
    acc[1][0] = WMMA_F16(a1.v, b0.v, acc[1][0]);
    acc[1][1] = WMMA_F16(a1.v, b1.v, acc[1][1]);
  }

  for (int i = 0; i < 2; ++i)
    for (int j = 0; j < 2; ++j)
      for (int e = 0; e < 8; ++e) {
        int row = row0 + i * 16 + half * 8 + e;
        int col = col0 + j * 16 + r;
        if (row < M) {
          float v = acc[i][j][e] + (bias ? bias[col] : 0.f);
          if (relu) v = fmaxf(v, 0.f);
          if (Cf) Cf[(size_t)row * N + col] = v;
          if (Ch) Ch[(size_t)row * N + col] = (_Float16)v;
        }
      }
}

// ---------------- flash self-attention ----------------
// qk: [7200, 512] f16  (cols 0..255 = Q proj, 256..511 = K proj), v: [7200, 256] f16
// one wave per (b, h, 16-query tile); 4 waves per block.

__global__ __launch_bounds__(128) void attn_kernel(const _Float16* __restrict__ qk,
                                                   const _Float16* __restrict__ vbuf,
                                                   _Float16* __restrict__ outh) {
  __shared__ _Float16 plds[4][16 * 40];
  const int lane = threadIdx.x & 31;
  const int wid  = threadIdx.x >> 5;
  const int r = lane & 15, half = lane >> 4;
  const int QT = 57;                                  // ceil(900/16)
  int task = blockIdx.x * 4 + wid;
  int qt = task % QT;
  int rem = task / QT;
  int h = rem & 7;
  int b = rem >> 3;
  if (b >= Bb) return;

  Frag16 aq;
  {
    int q = qt * 16 + r; if (q > 899) q = 899;
    const _Float16* p = qk + (size_t)(q * Bb + b) * 512 + h * 32 + half * 8;
    aq.q[0] = *(const uint4*)p; aq.q[1] = *(const uint4*)(p + 16);
  }

  float mrow[8], lrow[8];
  v8f acc0 = {}, acc1 = {};
  for (int e = 0; e < 8; ++e) { mrow[e] = -1e30f; lrow[e] = 0.f; }
  _Float16* myp = plds[wid];
  const float scale = 0.17677669529663687f;           // 1/sqrt(32)

  for (int kt = 0; kt < 29; ++kt) {                   // 29*32 = 928 >= 900
    int tk = kt * 32;
    v8f s0, s1;
    {
      int tok = tk + r;
      int tokc = tok > 899 ? 899 : tok;
      Frag16 bk;
      const _Float16* p = qk + (size_t)(tokc * Bb + b) * 512 + 256 + h * 32 + half * 8;
      bk.q[0] = *(const uint4*)p; bk.q[1] = *(const uint4*)(p + 16);
      v8f z = {};
      s0 = WMMA_F16(aq.v, bk.v, z);
      float addm = (tok > 899) ? -1e30f : 0.f;
      for (int e = 0; e < 8; ++e) s0[e] = s0[e] * scale + addm;
    }
    {
      int tok = tk + 16 + r;
      int tokc = tok > 899 ? 899 : tok;
      Frag16 bk;
      const _Float16* p = qk + (size_t)(tokc * Bb + b) * 512 + 256 + h * 32 + half * 8;
      bk.q[0] = *(const uint4*)p; bk.q[1] = *(const uint4*)(p + 16);
      v8f z = {};
      s1 = WMMA_F16(aq.v, bk.v, z);
      float addm = (tok > 899) ? -1e30f : 0.f;
      for (int e = 0; e < 8; ++e) s1[e] = s1[e] * scale + addm;
    }

    // online softmax: C-fragment rows live in 16-lane groups -> width-16 shuffles
    float pr0[8], pr1[8];
    for (int e = 0; e < 8; ++e) {
      float tmx = fmaxf(s0[e], s1[e]);
      for (int msk = 1; msk < 16; msk <<= 1) tmx = fmaxf(tmx, __shfl_xor(tmx, msk, 16));
      float mnew = fmaxf(mrow[e], tmx);
      float rs = __expf(mrow[e] - mnew);
      float p0 = __expf(s0[e] - mnew);
      float p1 = __expf(s1[e] - mnew);
      float psum = p0 + p1;
      for (int msk = 1; msk < 16; msk <<= 1) psum += __shfl_xor(psum, msk, 16);
      lrow[e] = lrow[e] * rs + psum;
      mrow[e] = mnew;
      acc0[e] *= rs; acc1[e] *= rs;
      pr0[e] = p0; pr1[e] = p1;
    }

    // stage P (16x32) to LDS in row-major, then re-read as A fragment
    for (int e = 0; e < 8; ++e) {
      int rowi = half * 8 + e;
      myp[rowi * 40 + r]      = (_Float16)pr0[e];
      myp[rowi * 40 + 16 + r] = (_Float16)pr1[e];
    }
    Frag16 pa;
    {
      const unsigned* p  = (const unsigned*)(myp + r * 40 + half * 8);
      const unsigned* p2 = (const unsigned*)(myp + r * 40 + 16 + half * 8);
      pa.u[0] = p[0];  pa.u[1] = p[1];  pa.u[2] = p[2];  pa.u[3] = p[3];
      pa.u[4] = p2[0]; pa.u[5] = p2[1]; pa.u[6] = p2[2]; pa.u[7] = p2[3];
    }

    // O += P @ V   (two N=16 halves of DH=32)
    for (int j = 0; j < 2; ++j) {
      Frag16 bv;
      int dcol = j * 16 + r;
      for (int i2 = 0; i2 < 16; ++i2) {
        int kk = (i2 < 8) ? (half * 8 + i2) : (16 + half * 8 + i2 - 8);
        int tok = tk + kk; if (tok > 899) tok = 899;
        bv.h[i2] = vbuf[(size_t)(tok * Bb + b) * 256 + h * 32 + dcol];
      }
      if (j == 0) acc0 = WMMA_F16(pa.v, bv.v, acc0);
      else        acc1 = WMMA_F16(pa.v, bv.v, acc1);
    }
  }

  for (int e = 0; e < 8; ++e) {
    int q = qt * 16 + half * 8 + e;
    if (q < 900) {
      float inv = 1.f / lrow[e];
      size_t base = (size_t)(q * Bb + b) * 256 + h * 32;
      outh[base + r]      = (_Float16)(acc0[e] * inv);
      outh[base + 16 + r] = (_Float16)(acc1[e] * inv);
    }
  }
}

// ---------------- deformable sampling ----------------
// one wave per (b, nq, h); lane = channel dh. value: [B*S, 256] f16.

__global__ __launch_bounds__(256) void deform_kernel(const _Float16* __restrict__ val,
                                                     const float* __restrict__ off,
                                                     const float* __restrict__ aw,
                                                     const float* __restrict__ ref,
                                                     _Float16* __restrict__ outh) {
  const int lane = threadIdx.x & 31;
  int task = blockIdx.x * 8 + (threadIdx.x >> 5);
  int h  = task & 7;
  int nq = (task >> 3) % 900;
  int b  = task / (8 * 900);
  if (b >= Bb) return;
  int m = nq * Bb + b;

  // softmax over L*P = 20 attention logits (redundant across lanes; L2-hot)
  float w[20]; float mx = -1e30f;
  const float* awp = aw + (size_t)m * 160 + h * 20;
  for (int i = 0; i < 20; ++i) { w[i] = awp[i]; mx = fmaxf(mx, w[i]); }
  float ssum = 0.f;
  for (int i = 0; i < 20; ++i) { w[i] = __expf(w[i] - mx); ssum += w[i]; }
  float rinv = 1.f / ssum;

  const int HL[4] = {128, 64, 32, 16};
  const int ST[4] = {0, 16384, 20480, 21504};
  const float* offp = off + (size_t)m * 320;
  float acc = 0.f;
  for (int l = 0; l < 4; ++l) {
    int Wl = HL[l], Hl = HL[l];
    const float* rp = ref + ((size_t)(nq * Bb + b) * 4 + l) * 4;   // [NQ,B,L,4], L==4
    float cx = rp[0], cy = rp[1], cw = rp[2], chh = rp[3];
    for (int p5 = 0; p5 < 5; ++p5) {
      int oc = ((h * 4 + l) * 5 + p5) * 2;
      float ox = offp[oc], oy = offp[oc + 1];
      float lx = cx + ox * 0.2f * cw * 0.5f;
      float ly = cy + oy * 0.2f * chh * 0.5f;
      float x = lx * Wl - 0.5f, y = ly * Hl - 0.5f;
      float x0 = floorf(x), y0 = floorf(y);
      float fx = x - x0, fy = y - y0;
      int x0i = (int)x0, y0i = (int)y0;
      float ws = w[l * 5 + p5] * rinv;
      float samp = 0.f;
      for (int dy = 0; dy < 2; ++dy)
        for (int dx = 0; dx < 2; ++dx) {
          int xi = x0i + dx, yi = y0i + dy;
          if (xi >= 0 && xi < Wl && yi >= 0 && yi < Hl) {
            float wbi = (dx ? fx : 1.f - fx) * (dy ? fy : 1.f - fy);
            size_t vidx = ((size_t)b * Ss + ST[l] + (size_t)yi * Wl + xi) * 256 + h * 32 + lane;
            samp += wbi * (float)val[vidx];
          }
        }
      acc += ws * samp;
    }
  }
  outh[(size_t)m * 256 + h * 32 + lane] = (_Float16)acc;
}

// ---------------- residual + layernorm ----------------
// one wave per row; outputs f32 residual stream, f16 for next GEMM, optional (y+pos) f16.

__global__ __launch_bounds__(256) void resln_kernel(const float* __restrict__ a,
                                                    const float* __restrict__ c,
                                                    const float* __restrict__ g,
                                                    const float* __restrict__ bt,
                                                    const float* __restrict__ pos,
                                                    float* __restrict__ y32,
                                                    _Float16* __restrict__ y16,
                                                    _Float16* __restrict__ q16, int Mr) {
  const int lane = threadIdx.x & 31;
  int m = blockIdx.x * 8 + (threadIdx.x >> 5);
  if (m >= Mr) return;
  float x[8];
  const float* ap = a + (size_t)m * 256 + lane * 8;
  const float* cp = c + (size_t)m * 256 + lane * 8;
  float s = 0.f, s2 = 0.f;
  for (int j = 0; j < 8; ++j) { x[j] = ap[j] + cp[j]; s += x[j]; s2 += x[j] * x[j]; }
  for (int msk = 1; msk < 32; msk <<= 1) { s += __shfl_xor(s, msk, 32); s2 += __shfl_xor(s2, msk, 32); }
  float mean = s * (1.f / 256.f);
  float var  = s2 * (1.f / 256.f) - mean * mean;
  float rstd = rsqrtf(var + 1e-5f);
  for (int j = 0; j < 8; ++j) {
    int col = lane * 8 + j;
    float y = (x[j] - mean) * rstd * g[col] + bt[col];
    if (y32) y32[(size_t)m * 256 + col] = y;
    if (y16) y16[(size_t)m * 256 + col] = (_Float16)y;
    if (q16) q16[(size_t)m * 256 + col] = (_Float16)(y + pos[(size_t)m * 256 + col]);
  }
}

// ---------------- host orchestration ----------------

static inline dim3 gemm_grid(int M, int N) { return dim3((M + 127) / 128, (N + 63) / 64); }

extern "C" void kernel_launch(void* const* d_in, const int* in_sizes, int n_in,
                              void* d_out, int out_size, void* d_ws, size_t ws_size,
                              hipStream_t stream) {
  (void)in_sizes; (void)n_in; (void)out_size; (void)ws_size;
  const float* tgt      = (const float*)d_in[0];
  const float* pos      = (const float*)d_in[1];
  const float* ref      = (const float*)d_in[2];
  const float* mem      = (const float*)d_in[3];
  const float* sa_in_w  = (const float*)d_in[4];
  const float* sa_in_b  = (const float*)d_in[5];
  const float* sa_out_w = (const float*)d_in[6];
  const float* sa_out_b = (const float*)d_in[7];
  const float* off_w    = (const float*)d_in[8];
  const float* off_b    = (const float*)d_in[9];
  const float* aw_w     = (const float*)d_in[10];
  const float* aw_b     = (const float*)d_in[11];
  const float* val_w    = (const float*)d_in[12];
  const float* val_b    = (const float*)d_in[13];
  const float* co_w     = (const float*)d_in[14];
  const float* co_b     = (const float*)d_in[15];
  const float* w1       = (const float*)d_in[16];
  const float* b1       = (const float*)d_in[17];
  const float* w2       = (const float*)d_in[18];
  const float* b2       = (const float*)d_in[19];
  const float* ln1_g    = (const float*)d_in[20];
  const float* ln1_b    = (const float*)d_in[21];
  const float* ln2_g    = (const float*)d_in[22];
  const float* ln2_b    = (const float*)d_in[23];
  const float* ln3_g    = (const float*)d_in[24];
  const float* ln3_b    = (const float*)d_in[25];

  char* ws = (char*)d_ws;
  size_t cur = 0;
  auto alloc = [&](size_t bytes) -> char* {
    char* p = ws + cur;
    cur += (bytes + 255) & ~(size_t)255;
    return p;
  };

  const size_t MD = (size_t)Mrows * 256;
  _Float16* q_h     = (_Float16*)alloc(MD * 2);                 // (tgt+pos) f16
  _Float16* tgt_h   = (_Float16*)alloc(MD * 2);                 // tgt f16
  _Float16* qk_h    = (_Float16*)alloc((size_t)Mrows * 512 * 2);// q|k proj f16
  _Float16* v_h     = (_Float16*)alloc(MD * 2);                 // v proj f16
  _Float16* attn_h  = (_Float16*)alloc(MD * 2);                 // attention output f16
  float*    sa_f    = (float*)alloc(MD * 4);                    // self-attn out proj f32
  float*    tgt2_f  = (float*)alloc(MD * 4);                    // after LN2 f32
  _Float16* q2_h    = (_Float16*)alloc(MD * 2);                 // (tgt2+pos) f16
  _Float16* mem_h   = (_Float16*)alloc((size_t)Bb * Ss * 256 * 2);
  _Float16* val_h   = (_Float16*)alloc((size_t)Bb * Ss * 256 * 2);
  float*    off_f   = (float*)alloc((size_t)Mrows * 320 * 4);
  float*    aw_f    = (float*)alloc((size_t)Mrows * 160 * 4);
  _Float16* samp_h  = (_Float16*)alloc(MD * 2);                 // deformable output f16
  float*    ca_f    = (float*)alloc(MD * 4);                    // cross-attn out proj f32
  float*    tgt3_f  = (float*)alloc(MD * 4);                    // after LN1 f32
  _Float16* tgt3_h  = (_Float16*)alloc(MD * 2);
  _Float16* ffh_h   = (_Float16*)alloc((size_t)Mrows * 1024 * 2);
  float*    ff2_f   = (float*)alloc(MD * 4);

  _Float16* sa_in_w_h  = (_Float16*)alloc((size_t)768 * 256 * 2);
  _Float16* sa_out_w_h = (_Float16*)alloc((size_t)256 * 256 * 2);
  _Float16* off_w_h    = (_Float16*)alloc((size_t)320 * 256 * 2);
  _Float16* aw_w_h     = (_Float16*)alloc((size_t)160 * 256 * 2);
  _Float16* val_w_h    = (_Float16*)alloc((size_t)256 * 256 * 2);
  _Float16* co_w_h     = (_Float16*)alloc((size_t)256 * 256 * 2);
  _Float16* w1_h       = (_Float16*)alloc((size_t)1024 * 256 * 2);
  _Float16* w2_h       = (_Float16*)alloc((size_t)256 * 1024 * 2);

  auto cvt = [&](const float* s, _Float16* d, int n) {
    cvt_kernel<<<(n + 255) / 256, 256, 0, stream>>>(s, d, n);
  };

  // --- conversions ---
  pack_q_kernel<<<((int)MD + 255) / 256, 256, 0, stream>>>(tgt, pos, q_h, tgt_h, (int)MD);
  cvt(sa_in_w, sa_in_w_h, 768 * 256);
  cvt(sa_out_w, sa_out_w_h, 256 * 256);
  cvt(off_w, off_w_h, 320 * 256);
  cvt(aw_w, aw_w_h, 160 * 256);
  cvt(val_w, val_w_h, 256 * 256);
  cvt(co_w, co_w_h, 256 * 256);
  cvt(w1, w1_h, 1024 * 256);
  cvt(w2, w2_h, 256 * 1024);
  {
    int n = Ss * Bb * 64;
    memtr_kernel<<<(n + 255) / 256, 256, 0, stream>>>(mem, mem_h);
  }

  // --- self-attention ---
  gemm_f16<<<gemm_grid(Mrows, 512), 256, 0, stream>>>(q_h, sa_in_w_h, sa_in_b,
                                                      nullptr, qk_h, Mrows, 512, 256, 0);
  gemm_f16<<<gemm_grid(Mrows, 256), 256, 0, stream>>>(tgt_h, sa_in_w_h + (size_t)512 * 256,
                                                      sa_in_b + 512, nullptr, v_h,
                                                      Mrows, 256, 256, 0);
  attn_kernel<<<(Bb * Hh * 57) / 4, 128, 0, stream>>>(qk_h, v_h, attn_h);
  gemm_f16<<<gemm_grid(Mrows, 256), 256, 0, stream>>>(attn_h, sa_out_w_h, sa_out_b,
                                                      sa_f, nullptr, Mrows, 256, 256, 0);
  resln_kernel<<<Mrows / 8, 256, 0, stream>>>(tgt, sa_f, ln2_g, ln2_b, pos,
                                              tgt2_f, nullptr, q2_h, Mrows);

  // --- deformable cross-attention ---
  gemm_f16<<<gemm_grid(Bb * Ss, 256), 256, 0, stream>>>(mem_h, val_w_h, val_b,
                                                        nullptr, val_h, Bb * Ss, 256, 256, 0);
  gemm_f16<<<gemm_grid(Mrows, 320), 256, 0, stream>>>(q2_h, off_w_h, off_b,
                                                      off_f, nullptr, Mrows, 320, 256, 0);
  gemm_f16<<<gemm_grid(Mrows, 160), 256, 0, stream>>>(q2_h, aw_w_h, aw_b,
                                                      aw_f, nullptr, Mrows, 160, 256, 0);
  deform_kernel<<<(Bb * NQq * Hh) / 8, 256, 0, stream>>>(val_h, off_f, aw_f, ref, samp_h);
  gemm_f16<<<gemm_grid(Mrows, 256), 256, 0, stream>>>(samp_h, co_w_h, co_b,
                                                      ca_f, nullptr, Mrows, 256, 256, 0);
  resln_kernel<<<Mrows / 8, 256, 0, stream>>>(tgt2_f, ca_f, ln1_g, ln1_b, nullptr,
                                              tgt3_f, tgt3_h, nullptr, Mrows);

  // --- FFN ---
  gemm_f16<<<gemm_grid(Mrows, 1024), 256, 0, stream>>>(tgt3_h, w1_h, b1,
                                                       nullptr, ffh_h, Mrows, 1024, 256, 1);
  gemm_f16<<<gemm_grid(Mrows, 256), 256, 0, stream>>>(ffh_h, w2_h, b2,
                                                      ff2_f, nullptr, Mrows, 256, 1024, 0);
  resln_kernel<<<Mrows / 8, 256, 0, stream>>>(tgt3_f, ff2_f, ln3_g, ln3_b, nullptr,
                                              (float*)d_out, nullptr, nullptr, Mrows);
}
